// LowRankTokenMix_43748536877633
// MI455X (gfx1250) — matrix operand order repaired and checked by reading, not compiled, not run on hardware
//
#include <hip/hip_runtime.h>

#define DIM   512
#define RANK  64
#define NTOK  4096
#define NBATCH 4
#define LOG2E 1.44269504088896340736f

typedef __attribute__((ext_vector_type(16))) __bf16 v16bf;
typedef __attribute__((ext_vector_type(8)))  float  v8f;

union BF16x16 {
    v16bf v;
    unsigned int u[8];
    unsigned short h[16];
};

static __device__ __forceinline__ unsigned short to_bf16u(float a) {
    union { __bf16 f; unsigned short u; } t;
    t.f = (__bf16)a;
    return t.u;
}
static __device__ __forceinline__ unsigned int pack2_bf16(float a, float b) {
#if __has_builtin(__builtin_amdgcn_cvt_pk_bf16_f32)
    auto r = __builtin_amdgcn_cvt_pk_bf16_f32(a, b);   // single v_cvt_pk_bf16_f32
    unsigned int u;
    __builtin_memcpy(&u, &r, sizeof(u));
    return u;
#else
    union { __bf16 f[2]; unsigned int u; } t;
    t.f[0] = (__bf16)a;
    t.f[1] = (__bf16)b;
    return t.u;
#endif
}

// ---------------------------------------------------------------------------
// Kernel 1: b = x @ W_B^T (fp32 in, bf16 out) + per-row L2 norm of b.
// ---------------------------------------------------------------------------
__global__ __launch_bounds__(256) void bproj_kernel(const float* __restrict__ x,
                                                    const float* __restrict__ wb,
                                                    unsigned short* __restrict__ b_bf16,
                                                    float* __restrict__ norms) {
    __shared__ float xs[16 * DIM];   // 32 KB
    const int tid  = threadIdx.x;
    const int row0 = blockIdx.x * 16;          // flat over 4*4096 rows

    const float4* xsrc = (const float4*)(x + (size_t)row0 * DIM);
    float4* xd = (float4*)xs;
#pragma unroll
    for (int i = 0; i < 8; ++i) xd[tid + 256 * i] = xsrc[tid + 256 * i];
    __syncthreads();

    const int r  = tid >> 4;        // 0..15 row in tile
    const int cb = tid & 15;        // base output column
    float acc[4] = {0.f, 0.f, 0.f, 0.f};
    const float4* xr  = (const float4*)(xs + r * DIM);
    const float4* wb4 = (const float4*)wb;
    for (int k4 = 0; k4 < DIM / 4; ++k4) {
        float4 xv = xr[k4];
#pragma unroll
        for (int j = 0; j < 4; ++j) {
            float4 wv = wb4[(size_t)(cb + 16 * j) * (DIM / 4) + k4];
            acc[j] += xv.x * wv.x + xv.y * wv.y + xv.z * wv.z + xv.w * wv.w;
        }
    }
    unsigned short* dst = b_bf16 + (size_t)(row0 + r) * RANK;
#pragma unroll
    for (int j = 0; j < 4; ++j) dst[cb + 16 * j] = to_bf16u(acc[j]);

    float ss = acc[0] * acc[0] + acc[1] * acc[1] + acc[2] * acc[2] + acc[3] * acc[3];
    ss += __shfl_xor(ss, 1);
    ss += __shfl_xor(ss, 2);
    ss += __shfl_xor(ss, 4);
    ss += __shfl_xor(ss, 8);
    if (cb == 0) norms[row0 + r] = __builtin_sqrtf(ss);
}

// ---------------------------------------------------------------------------
// Kernel 2a: convert W_C [512][64] fp32 -> bf16
// ---------------------------------------------------------------------------
__global__ __launch_bounds__(256) void convw_kernel(const float* __restrict__ wc,
                                                    unsigned short* __restrict__ wcb) {
    int i = blockIdx.x * 256 + threadIdx.x;      // 32768 elements
    wcb[i] = to_bf16u(wc[i]);
}

// ---------------------------------------------------------------------------
// Kernel 2b: per-batch max row norm (one block per batch)
// ---------------------------------------------------------------------------
__global__ __launch_bounds__(256) void maxnorm_kernel(const float* __restrict__ norms,
                                                      float* __restrict__ mx) {
    __shared__ float s[256];
    const int tid = threadIdx.x;
    const int b   = blockIdx.x;
    float v = 0.0f;
    for (int i = tid; i < NTOK; i += 256) v = fmaxf(v, norms[(size_t)b * NTOK + i]);
    s[tid] = v;
    __syncthreads();
    for (int o = 128; o > 0; o >>= 1) {
        if (tid < o) s[tid] = fmaxf(s[tid], s[tid + o]);
        __syncthreads();
    }
    if (tid == 0) mx[b] = s[0];
}

// ---------------------------------------------------------------------------
// Kernel 3: fixed-shift flash attention over b (Q=K=V=b), bf16 WMMA.
// Key dimension split in 2 halves (grid doubled -> 2048 waves, >=2 waves/SIMD);
// halves write f32 partial (acc, l) that a combine kernel sums (deterministic).
// Row sums via a 5th WMMA against a constant ones-column B-frag.
// LDS strides padded (70/38/38) for conflict-free banking (64 x 32-bit banks).
// ---------------------------------------------------------------------------
#define WPB 4
#define ATTN_THREADS (WPB * 32)
#define BLOCKS_PER_BATCH ((NTOK / 16) / WPB)     // 64
#define KHALVES 2
#define KB_PER_HALF (NTOK / 32 / KHALVES)        // 64
#define KSTRIDE 70      // Kbuf row stride (elements), rows = key
#define TSTRIDE 38      // KTbuf row stride (elements), rows = rank
#define PSTRIDE 38      // Pbuf row stride (elements), rows = query

static __device__ __forceinline__ void stage_kblock(const unsigned short* __restrict__ bb,
                                                    int key0,
                                                    unsigned short* __restrict__ K,
                                                    unsigned short* __restrict__ KT,
                                                    int tid) {
    // 32 keys x 64 ranks bf16 = 256 uint4
    const uint4* src = (const uint4*)(bb + (size_t)key0 * RANK);
#pragma unroll
    for (int i = 0; i < 256 / ATTN_THREADS; ++i) {
        int q = i * ATTN_THREADS + tid;      // uint4 index 0..255
        uint4 d = src[q];
        int k  = q >> 3;                     // key row 0..31
        int e0 = (q & 7) * 8;                // first rank element of this uint4
        unsigned int* kr = (unsigned int*)(K + k * KSTRIDE + e0);
        kr[0] = d.x; kr[1] = d.y; kr[2] = d.z; kr[3] = d.w;
        // permuted transposed copy: key k -> column cidx = (k&15)*2 + (k>>4)
        int cidx = ((k & 15) << 1) | (k >> 4);
        unsigned short* kt = KT + (size_t)e0 * TSTRIDE + cidx;
        kt[0 * TSTRIDE] = (unsigned short)d.x; kt[1 * TSTRIDE] = (unsigned short)(d.x >> 16);
        kt[2 * TSTRIDE] = (unsigned short)d.y; kt[3 * TSTRIDE] = (unsigned short)(d.y >> 16);
        kt[4 * TSTRIDE] = (unsigned short)d.z; kt[5 * TSTRIDE] = (unsigned short)(d.z >> 16);
        kt[6 * TSTRIDE] = (unsigned short)d.w; kt[7 * TSTRIDE] = (unsigned short)(d.w >> 16);
    }
}

// B-fragment: pair (k,k+1) at base[((v < 4) ? 2v : 2v + 8) + 8h]
static __device__ __forceinline__ v16bf load_bfrag(const unsigned short* base, int h) {
    BF16x16 f;
#pragma unroll
    for (int v = 0; v < 8; ++v) f.u[v] = *(const unsigned int*)(base + ((v < 4) ? 2 * v : 2 * v + 8) + 8 * h);
    return f.v;
}

#define WMMA_BF16(A, B, C) \
    __builtin_amdgcn_wmma_f32_16x16x32_bf16(false, (A), false, (B), (short)0, (C), false, false)

__global__ __launch_bounds__(ATTN_THREADS) void attn_kernel(const unsigned short* __restrict__ b_bf16,
                                                            const float* __restrict__ norms,
                                                            const float* __restrict__ maxnorm,
                                                            float* __restrict__ pacc,
                                                            float* __restrict__ pl) {
    __shared__ unsigned short Kbuf[2][32 * KSTRIDE];
    __shared__ unsigned short KTbuf[2][RANK * TSTRIDE];   // key-permuted transpose
    __shared__ unsigned short Pbuf[WPB][16 * PSTRIDE];    // key-permuted P

    const int tid   = threadIdx.x;
    const int lane  = tid & 31;
    const int wave  = tid >> 5;
    const int half  = blockIdx.x & (KHALVES - 1);
    const int rest  = blockIdx.x / KHALVES;
    const int batch = rest / BLOCKS_PER_BATCH;
    const int qtile = (rest % BLOCKS_PER_BATCH) * WPB + wave;
    const int qrow0 = qtile * 16;
    const unsigned short* bb = b_bf16 + (size_t)batch * NTOK * RANK;

    const int m = lane & 15;   // A-frag row / B-frag column / D-frag column
    const int h = lane >> 4;   // lane half

    // Q A-fragments (16x64 -> two 16x32 frags)
    BF16x16 qa0, qa1;
#pragma unroll
    for (int v = 0; v < 8; ++v) {
        int kb = ((v < 4) ? 2 * v : 2 * v + 8) + 8 * h;   // A-frag K index
        const unsigned int* row = (const unsigned int*)(bb + (size_t)(qrow0 + m) * RANK);
        qa0.u[v] = row[kb >> 1];
        qa1.u[v] = row[(kb + 32) >> 1];
    }

    // fixed per-row softmax shift, pre-scaled: p = exp2(s*LOG2E + MvL)
    float MvL[8];
    {
        const float mn = maxnorm[batch];
        const float* nr = norms + (size_t)batch * NTOK + qrow0;
#pragma unroll
        for (int v = 0; v < 8; ++v) MvL[v] = -(nr[v + 8 * h] * mn) * LOG2E;
    }

    // constant B-fragment: ones in column 0 (row-sum accumulator column)
    BF16x16 ones;
#pragma unroll
    for (int v = 0; v < 8; ++v) ones.u[v] = (m == 0) ? 0x3F803F80u : 0u;

    v8f acc0 = {}, acc1 = {}, acc2 = {}, acc3 = {}, acc4 = {};

    const int kb0   = half * KB_PER_HALF;
    const int kbend = kb0 + KB_PER_HALF;

    stage_kblock(bb, kb0 * 32, Kbuf[0], KTbuf[0], tid);
    __syncthreads();

    for (int kb = kb0; kb < kbend; ++kb) {
        const int cur = kb & 1;
        if (kb + 1 < kbend)
            stage_kblock(bb, (kb + 1) * 32, Kbuf[1 - cur], KTbuf[1 - cur], tid);

        const unsigned short* K  = Kbuf[cur];
        const unsigned short* KT = KTbuf[cur];

        // ---- S = Q x K^T : 16x32 scores as two 16x16 D tiles ----
        v8f s0 = {}, s1 = {};
        s0 = WMMA_BF16(qa0.v, load_bfrag(K + m * KSTRIDE, h), s0);
        s0 = WMMA_BF16(qa1.v, load_bfrag(K + m * KSTRIDE + 32, h), s0);
        s1 = WMMA_BF16(qa0.v, load_bfrag(K + (16 + m) * KSTRIDE, h), s1);
        s1 = WMMA_BF16(qa1.v, load_bfrag(K + (16 + m) * KSTRIDE + 32, h), s1);

        // ---- P = exp2(S*log2e - M*log2e): fixed shift, raw v_exp_f32 ----
        // packed store under key permutation pi(2n)=n, pi(2n+1)=n+16
        unsigned short* P = Pbuf[wave];
#pragma unroll
        for (int v = 0; v < 8; ++v) {
            float p0 = __builtin_amdgcn_exp2f(__builtin_fmaf(s0[v], LOG2E, MvL[v]));
            float p1 = __builtin_amdgcn_exp2f(__builtin_fmaf(s1[v], LOG2E, MvL[v]));
            *(unsigned int*)(P + (v + 8 * h) * PSTRIDE + 2 * m) = pack2_bf16(p0, p1);
        }
        asm volatile("" ::: "memory");   // keep DS store->load order
        BF16x16 pa;
#pragma unroll
        for (int v = 0; v < 8; ++v) {
            int kk = ((v < 4) ? 2 * v : 2 * v + 8) + 8 * h;
            pa.u[v] = *(const unsigned int*)(P + m * PSTRIDE + kk);
        }

        // ---- acc += P (16x32) x V (32x64) + row-sum column ----
        acc0 = WMMA_BF16(pa.v, load_bfrag(KT + (m)      * TSTRIDE, h), acc0);
        acc1 = WMMA_BF16(pa.v, load_bfrag(KT + (16 + m) * TSTRIDE, h), acc1);
        acc2 = WMMA_BF16(pa.v, load_bfrag(KT + (32 + m) * TSTRIDE, h), acc2);
        acc3 = WMMA_BF16(pa.v, load_bfrag(KT + (48 + m) * TSTRIDE, h), acc3);
        acc4 = WMMA_BF16(pa.v, ones.v, acc4);   // row sums -> column 0

        __syncthreads();
    }

    // ---- write f32 partials (deterministic two-pass combine) ----
    float* pa_out = pacc + ((size_t)half * NBATCH * NTOK + (size_t)batch * NTOK) * RANK;
#pragma unroll
    for (int v = 0; v < 8; ++v) {
        float* dst = pa_out + (size_t)(qrow0 + v + 8 * h) * RANK;
        dst[m]      = acc0[v];
        dst[16 + m] = acc1[v];
        dst[32 + m] = acc2[v];
        dst[48 + m] = acc3[v];
    }
    if (m == 0) {
        float* pld = pl + (size_t)half * NBATCH * NTOK + (size_t)batch * NTOK + qrow0;
#pragma unroll
        for (int v = 0; v < 8; ++v) pld[v + 8 * h] = acc4[v];
    }
}

// ---------------------------------------------------------------------------
// Kernel 3b: combine key-halves, normalize, emit mixed bf16
// ---------------------------------------------------------------------------
__global__ __launch_bounds__(256) void combine_kernel(const float* __restrict__ pacc,
                                                      const float* __restrict__ pl,
                                                      unsigned short* __restrict__ mixed) {
    const size_t HALF = (size_t)NBATCH * NTOK * RANK;
    int i = blockIdx.x * 256 + threadIdx.x;     // over NBATCH*NTOK*RANK
    int row = i >> 6;                           // / RANK
    float a = pacc[i] + pacc[HALF + i];
    float l = pl[row] + pl[NBATCH * NTOK + row];
    mixed[i] = to_bf16u(a / l);
}

// ---------------------------------------------------------------------------
// Kernel 4: out = mixed @ W_C^T  (WMMA bf16, f32 out).  4 waves / block.
// ---------------------------------------------------------------------------
__global__ __launch_bounds__(128) void proj_kernel(const unsigned short* __restrict__ mixed,
                                                   const unsigned short* __restrict__ wcb,
                                                   float* __restrict__ out) {
    const int lane = threadIdx.x & 31;
    const int wave = threadIdx.x >> 5;
    const int tile = blockIdx.x * 4 + wave;   // flat over 16384 rows / 16
    const int row0 = tile * 16;
    const int m = lane & 15;
    const int h = lane >> 4;

    BF16x16 a0, a1;
#pragma unroll
    for (int v = 0; v < 8; ++v) {
        int kb = ((v < 4) ? 2 * v : 2 * v + 8) + 8 * h;
        const unsigned int* row = (const unsigned int*)(mixed + (size_t)(row0 + m) * RANK);
        a0.u[v] = row[kb >> 1];
        a1.u[v] = row[(kb + 32) >> 1];
    }

    for (int c = 0; c < DIM / 16; ++c) {
        const unsigned short* wrow = wcb + (size_t)(16 * c + m) * RANK;
        BF16x16 f0, f1;
#pragma unroll
        for (int v = 0; v < 8; ++v) {
            f0.u[v] = *(const unsigned int*)(wrow + ((v < 4) ? 2 * v : 2 * v + 8) + 8 * h);
            f1.u[v] = *(const unsigned int*)(wrow + ((v < 4) ? 2 * v : 2 * v + 8) + 8 * h + 32);
        }
        v8f d = {};
        d = WMMA_BF16(a0.v, f0.v, d);
        d = WMMA_BF16(a1.v, f1.v, d);
#pragma unroll
        for (int v = 0; v < 8; ++v)
            out[(size_t)(row0 + v + 8 * h) * DIM + 16 * c + m] = d[v];
    }
}

// ---------------------------------------------------------------------------
extern "C" void kernel_launch(void* const* d_in, const int* in_sizes, int n_in,
                              void* d_out, int out_size, void* d_ws, size_t ws_size,
                              hipStream_t stream) {
    const float* x   = (const float*)d_in[0];   // [4,4096,512]
    const float* W_B = (const float*)d_in[1];   // [64,512]
    const float* W_C = (const float*)d_in[2];   // [512,64]
    float* out = (float*)d_out;                 // [4,4096,512]

    unsigned short* b_bf16 = (unsigned short*)d_ws;                 // 2 MB
    unsigned short* mixed  = b_bf16 + (size_t)NBATCH * NTOK * RANK; // 2 MB
    unsigned short* wcb    = mixed  + (size_t)NBATCH * NTOK * RANK; // 64 KB
    float* norms   = (float*)(wcb + (size_t)DIM * RANK);            // 64 KB
    float* mxnorm  = norms + (size_t)NBATCH * NTOK;                 // 16 B
    float* pacc    = mxnorm + NBATCH;                               // 8 MB (2 halves)
    float* pl      = pacc + (size_t)KHALVES * NBATCH * NTOK * RANK; // 128 KB

    bproj_kernel<<<(NBATCH * NTOK) / 16, 256, 0, stream>>>(x, W_B, b_bf16, norms);
    convw_kernel<<<(DIM * RANK) / 256, 256, 0, stream>>>(W_C, wcb);
    maxnorm_kernel<<<NBATCH, 256, 0, stream>>>(norms, mxnorm);
    attn_kernel<<<NBATCH * BLOCKS_PER_BATCH * KHALVES, ATTN_THREADS, 0, stream>>>(
        b_bf16, norms, mxnorm, pacc, pl);
    combine_kernel<<<(NBATCH * NTOK * RANK) / 256, 256, 0, stream>>>(pacc, pl, mixed);
    proj_kernel<<<(NBATCH * NTOK) / (16 * 4), 128, 0, stream>>>(mixed, wcb, out);
}
